// GRUD_V2_47725676593189
// MI455X (gfx1250) — compile-verified, matrix operand order checked by reading, not run on hardware
//
#include <hip/hip_runtime.h>
#include <hip/hip_bf16.h>

// GRU-D persistent-scan kernel for gfx1250 (MI455X).
// B=256, T=512, D=64, H=128. 16 workgroups x 256 threads (8 wave32 waves).
// Each WG owns 16 batch rows; h state lives in per-lane registers (WMMA D
// fragment pattern); bf16-swizzled weights live in LDS; scan over 512 steps.
// Per step per wave: 26 v_wmma_f32_16x16x32_bf16; 3 workgroup barriers.
// f32->bf16 conversions use v_fma_mixlo/mixhi_bf16 (CDNA5 VOP3P ops 62/63).

#define BB   256
#define TT   512
#define DD   64
#define HH   128
#define KC   256           // D + H + D
#define ROWS 16            // batch rows per workgroup

typedef __attribute__((ext_vector_type(16))) __bf16 v16bf;
typedef __attribute__((ext_vector_type(8)))  float  v8f;
typedef __attribute__((ext_vector_type(4)))  int    i4;
typedef __attribute__((ext_vector_type(8)))  int    i8;
typedef __attribute__((ext_vector_type(4)))  float  f4;

// f32 -> bf16 (RNE) into [15:0] via V_FMA_MIXLO_BF16 (x*1.0+0).
__device__ __forceinline__ unsigned short f2bf(float a) {
    unsigned d;
    asm("v_fma_mixlo_bf16 %0, %1, 1.0, 0" : "=v"(d) : "v"(a));
    return (unsigned short)d;
}

// two f32 -> packed 2xbf16 dword via MIXLO + MIXHI.
__device__ __forceinline__ unsigned packbf2(float a, float b) {
    unsigned d;
    asm("v_fma_mixlo_bf16 %0, %1, 1.0, 0" : "=v"(d) : "v"(a));
    asm("v_fma_mixhi_bf16 %0, %1, 1.0, 0" : "+v"(d) : "v"(b));
    return d;
}

__device__ __forceinline__ float sigmoid_fast(float x) {
    return __builtin_amdgcn_rcpf(1.0f + __expf(-x));
}
__device__ __forceinline__ float tanh_fast(float x) {
    // tanh(x) = 1 - 2/(exp(2x)+1); saturates correctly at +-1
    return 1.0f - 2.0f * __builtin_amdgcn_rcpf(__expf(2.0f * x) + 1.0f);
}

// A-fragment (16x32 bf16) from row-major bf16 LDS tile.
// ISA 7.12.2: lanes 0-15 hold K=0..7 (v0-3) and K=16..23 (v4-7); lanes 16-31 +8.
__device__ __forceinline__ v16bf load_a_frag(const unsigned short* base, int stride,
                                             int lane, int kt) {
    const int half = lane >> 4, m = lane & 15;
    const unsigned short* p = base + m * stride + kt * 32 + half * 8;
    i4 lo = *(const i4*)(p);
    i4 hi = *(const i4*)(p + 16);
    i8 w;
    w[0] = lo[0]; w[1] = lo[1]; w[2] = lo[2]; w[3] = lo[3];
    w[4] = hi[0]; w[5] = hi[1]; w[6] = hi[2]; w[7] = hi[3];
    return __builtin_bit_cast(v16bf, w);
}

// B-fragment from pre-swizzled LDS: [(tile)*32 + lane]*8 dwords, contiguous.
__device__ __forceinline__ v16bf load_b_frag(const unsigned int* p) {
    i4 lo = *(const i4*)(p);
    i4 hi = *(const i4*)(p + 4);
    i8 w;
    w[0] = lo[0]; w[1] = lo[1]; w[2] = lo[2]; w[3] = lo[3];
    w[4] = hi[0]; w[5] = hi[1]; w[6] = hi[2]; w[7] = hi[3];
    return __builtin_bit_cast(v16bf, w);
}

__device__ __forceinline__ v8f wmma_bf16(v16bf a, v16bf b, v8f c) {
    return __builtin_amdgcn_wmma_f32_16x16x32_bf16(false, a, false, b,
                                                   (short)0, c, false, false);
}

__global__ __launch_bounds__(256) void grud_scan_kernel(
    const float* __restrict__ input,   // [B,4,T,D]
    const float* __restrict__ W_gx,    // [D,D] (diag used)
    const float* __restrict__ b_gx,    // [D]
    const float* __restrict__ W_gh,    // [H,D]
    const float* __restrict__ b_gh,    // [H]
    const float* __restrict__ W_z,     // [H,256]
    const float* __restrict__ b_z,     // [H]
    const float* __restrict__ W_r,     // [H,256]
    const float* __restrict__ b_r,     // [H]
    const float* __restrict__ W_h,     // [H,256]
    const float* __restrict__ b_h,     // [H]
    float* __restrict__ out)           // [B,T,H]
{
    // ---- LDS (~222 KB) ----
    __shared__ __align__(16) unsigned int wz_sw[8 * 8 * 32 * 8];   // 64 KB
    __shared__ __align__(16) unsigned int wr_sw[8 * 8 * 32 * 8];   // 64 KB
    __shared__ __align__(16) unsigned int wh_sw[8 * 8 * 32 * 8];   // 64 KB
    __shared__ __align__(16) unsigned int wgh_sw[8 * 2 * 32 * 8];  // 16 KB
    __shared__ __align__(16) unsigned short combined[ROWS * KC];   // 8 KB (x|h|m)
    __shared__ __align__(16) unsigned short rh_lds[ROWS * HH];     // 4 KB (r*h)
    __shared__ __align__(16) unsigned short dlds[ROWS * DD];       // 2 KB

    const int tid  = threadIdx.x;
    const int wave = tid >> 5;
    const int lane = tid & 31;
    const int half = lane >> 4;
    const int b0   = blockIdx.x * ROWS;

    // ---- init: bf16-swizzle weights into WMMA B-fragment layout ----
    for (int idx = tid; idx < 8 * 8 * 32 * 8; idx += 256) {
        const int v  = idx & 7;
        const int ln = (idx >> 3) & 31;
        const int kt = (idx >> 8) & 7;
        const int nt = (idx >> 11) & 7;
        const int n  = nt * 16 + (ln & 15);
        const int k  = kt * 32 + ((v & 4) ? 16 : 0) + (ln >> 4) * 8 + (v & 3) * 2;
        wz_sw[idx] = packbf2(W_z[n * KC + k], W_z[n * KC + k + 1]);
        wr_sw[idx] = packbf2(W_r[n * KC + k], W_r[n * KC + k + 1]);
        wh_sw[idx] = packbf2(W_h[n * KC + k], W_h[n * KC + k + 1]);
    }
    for (int idx = tid; idx < 8 * 2 * 32 * 8; idx += 256) {
        const int v  = idx & 7;
        const int ln = (idx >> 3) & 31;
        const int kt = (idx >> 8) & 1;
        const int nt = (idx >> 9) & 7;
        const int n  = nt * 16 + (ln & 15);
        const int k  = kt * 32 + ((v & 4) ? 16 : 0) + (ln >> 4) * 8 + (v & 3) * 2;
        wgh_sw[idx] = packbf2(W_gh[n * DD + k], W_gh[n * DD + k + 1]);
    }

    // ---- loop-invariant per-lane scalars (registers, not LDS) ----
    const int nglb = wave * 16 + (lane & 15);   // output column this lane owns
    const float rb_gh = b_gh[nglb];
    const float rb_z  = b_z[nglb];
    const float rb_r  = b_r[nglb];
    const float rb_h  = b_h[nglb];

    const int r1  = tid >> 4;                   // phase-1 row this thread owns
    const int c0  = (tid & 15) * 4;             // phase-1 col group
    float dg[4], bg[4];
#pragma unroll
    for (int j = 0; j < 4; ++j) {
        dg[j] = W_gx[(c0 + j) * DD + (c0 + j)]; // diagonal of masked FilterLinear
        bg[j] = b_gx[c0 + j];
    }

    float hreg[8];                              // h state: D-fragment layout
#pragma unroll
    for (int v = 0; v < 8; ++v) hreg[v] = 0.0f;

    unsigned* const comb32 = (unsigned*)combined;
    unsigned* const d32    = (unsigned*)dlds;

    __syncthreads();

    // ---- software-pipelined input loads ----
    const size_t chs = (size_t)TT * DD;
    const float* in_base = input + (size_t)(b0 + r1) * 4 * chs + c0;
    f4 cx  = *(const f4*)(in_base + 0 * chs);
    f4 cxl = *(const f4*)(in_base + 1 * chs);
    f4 cm  = *(const f4*)(in_base + 2 * chs);
    f4 cd  = *(const f4*)(in_base + 3 * chs);

    for (int t = 0; t < TT; ++t) {
        // ---- phase 1: x_t, m, d tiles -> LDS (bf16, packed u32 stores) ----
        {
            float xv[4];
#pragma unroll
            for (int j = 0; j < 4; ++j) {
                const float delta_x = __expf(-fmaxf(0.0f, cd[j] * dg[j] + bg[j]));
                xv[j] = cm[j] * cx[j] + (1.0f - cm[j]) * (delta_x * cxl[j]);
            }
            const int cx32 = (r1 * KC + c0) >> 1;
            comb32[cx32 + 0]      = packbf2(xv[0], xv[1]);
            comb32[cx32 + 1]      = packbf2(xv[2], xv[3]);
            comb32[cx32 + 96 + 0] = packbf2(cm[0], cm[1]);   // +192 bf16 cols
            comb32[cx32 + 96 + 1] = packbf2(cm[2], cm[3]);
            const int cd32 = (r1 * DD + c0) >> 1;
            d32[cd32 + 0] = packbf2(cd[0], cd[1]);
            d32[cd32 + 1] = packbf2(cd[2], cd[3]);
        }
        // issue next step's loads now; they complete under phases 2-4
        f4 nx = cx, nxl = cxl, nm = cm, nd = cd;
        if (t + 1 < TT) {
            const float* nb = in_base + (size_t)(t + 1) * DD;
            nx  = *(const f4*)(nb + 0 * chs);
            nxl = *(const f4*)(nb + 1 * chs);
            nm  = *(const f4*)(nb + 2 * chs);
            nd  = *(const f4*)(nb + 3 * chs);
        }
        __syncthreads();                        // B1: tiles visible

        // ---- phase 2: delta_h GEMM [16,64]x[64,128]; h *= delta_h ----
        {
            v8f acc = {0.f, 0.f, 0.f, 0.f, 0.f, 0.f, 0.f, 0.f};
#pragma unroll
            for (int kt = 0; kt < 2; ++kt) {
                v16bf a = load_a_frag(dlds, DD, lane, kt);
                v16bf b = load_b_frag(wgh_sw + (((wave * 2 + kt) * 32 + lane) << 3));
                acc = wmma_bf16(a, b, acc);
            }
#pragma unroll
            for (int v = 0; v < 8; ++v) {
                const int M = v + 8 * half;
                const float dh = __expf(-fmaxf(0.0f, acc[v] + rb_gh));
                hreg[v] *= dh;
                combined[M * KC + 64 + nglb] = f2bf(hreg[v]);
            }
        }
        __syncthreads();                        // B2: combined (x|h|m) ready

        // ---- phase 3: z,r GEMMs [16,256]x[256,128] (shared A frags) ----
        float zf[8];
        v16bf ax0, ax1, am0, am1;               // x/m A-frags kept for phase 4
        {
            v8f accz = {0.f, 0.f, 0.f, 0.f, 0.f, 0.f, 0.f, 0.f};
            v8f accr = {0.f, 0.f, 0.f, 0.f, 0.f, 0.f, 0.f, 0.f};
#pragma unroll
            for (int kt = 0; kt < 8; ++kt) {
                v16bf a  = load_a_frag(combined, KC, lane, kt);
                if (kt == 0) ax0 = a;
                if (kt == 1) ax1 = a;
                if (kt == 6) am0 = a;
                if (kt == 7) am1 = a;
                v16bf bz = load_b_frag(wz_sw + (((wave * 8 + kt) * 32 + lane) << 3));
                v16bf br = load_b_frag(wr_sw + (((wave * 8 + kt) * 32 + lane) << 3));
                accz = wmma_bf16(a, bz, accz);
                accr = wmma_bf16(a, br, accr);
            }
#pragma unroll
            for (int v = 0; v < 8; ++v) {
                const int M = v + 8 * half;
                zf[v] = sigmoid_fast(accz[v] + rb_z);
                const float rv = sigmoid_fast(accr[v] + rb_r);
                rh_lds[M * HH + nglb] = f2bf(rv * hreg[v]);
            }
        }
        __syncthreads();                        // B3: rh ready (x/m reads done)

        // ---- phase 4: h_tilde GEMM; h = (1-z)*h + z*tanh(.) ; store ----
        {
            v8f acch = {0.f, 0.f, 0.f, 0.f, 0.f, 0.f, 0.f, 0.f};
            acch = wmma_bf16(ax0, load_b_frag(wh_sw + (((wave * 8 + 0) * 32 + lane) << 3)), acch);
            acch = wmma_bf16(ax1, load_b_frag(wh_sw + (((wave * 8 + 1) * 32 + lane) << 3)), acch);
#pragma unroll
            for (int kt = 0; kt < 4; ++kt) {    // h-section: K = 64..191
                v16bf a  = load_a_frag(rh_lds, HH, lane, kt);
                v16bf bh = load_b_frag(wh_sw + (((wave * 8 + 2 + kt) * 32 + lane) << 3));
                acch = wmma_bf16(a, bh, acch);
            }
            acch = wmma_bf16(am0, load_b_frag(wh_sw + (((wave * 8 + 6) * 32 + lane) << 3)), acch);
            acch = wmma_bf16(am1, load_b_frag(wh_sw + (((wave * 8 + 7) * 32 + lane) << 3)), acch);
#pragma unroll
            for (int v = 0; v < 8; ++v) {
                const int M = v + 8 * half;
                const float ht = tanh_fast(acch[v] + rb_h);
                const float hn = (1.0f - zf[v]) * hreg[v] + zf[v] * ht;
                hreg[v] = hn;
                out[((size_t)(b0 + M) * TT + t) * HH + nglb] = hn;
            }
        }
        // no end-of-step barrier needed: phase 4 touches only rh_lds (fenced by
        // next step's B1/B2 before any wave can rewrite it) and registers.

        cx = nx; cxl = nxl; cm = nm; cd = nd;
    }
}

extern "C" void kernel_launch(void* const* d_in, const int* in_sizes, int n_in,
                              void* d_out, int out_size, void* d_ws, size_t ws_size,
                              hipStream_t stream) {
    const float* input = (const float*)d_in[0];
    const float* W_gx  = (const float*)d_in[1];
    const float* b_gx  = (const float*)d_in[2];
    const float* W_gh  = (const float*)d_in[3];
    const float* b_gh  = (const float*)d_in[4];
    const float* W_z   = (const float*)d_in[5];
    const float* b_z   = (const float*)d_in[6];
    const float* W_r   = (const float*)d_in[7];
    const float* b_r   = (const float*)d_in[8];
    const float* W_h   = (const float*)d_in[9];
    const float* b_h   = (const float*)d_in[10];
    float* out = (float*)d_out;

    dim3 grid(BB / ROWS);   // 16 persistent workgroups, one per WGP
    dim3 block(256);        // 8 wave32 waves
    grud_scan_kernel<<<grid, block, 0, stream>>>(
        input, W_gx, b_gx, W_gh, b_gh, W_z, b_z, W_r, b_r, W_h, b_h, out);
}
